// CopyModel_82961588290052
// MI455X (gfx1250) — matrix operand order repaired and checked by reading, not compiled, not run on hardware
//
#include <hip/hip_runtime.h>
#include <hip/hip_bf16.h>
#include <math.h>

// ---------------- problem constants ----------------
#define BB 2
#define NN 4096
#define DD 512
#define HH 8
#define DH 64
#define FF 2048
#define CH 256           // chunk length for linear-attention scans
#define NC (NN / CH)     // 16 chunks
#define BND ((size_t)BB * NN * DD)

typedef __attribute__((ext_vector_type(16))) __bf16 v16bf;
typedef __attribute__((ext_vector_type(8)))  float  v8f;

// =====================================================================
// fp32 -> bf16 staging pass (8 elements / thread, exact grids)
// =====================================================================
__global__ __launch_bounds__(256)
void cvt_f32_bf16(const float* __restrict__ in, __bf16* __restrict__ out)
{
    const size_t i = ((size_t)blockIdx.x * 256 + threadIdx.x) * 8;
    const float4 f0 = ((const float4*)(in + i))[0];
    const float4 f1 = ((const float4*)(in + i))[1];
    union { uint4 u; __bf16 v[8]; } U;
    U.v[0] = (__bf16)f0.x; U.v[1] = (__bf16)f0.y; U.v[2] = (__bf16)f0.z; U.v[3] = (__bf16)f0.w;
    U.v[4] = (__bf16)f1.x; U.v[5] = (__bf16)f1.y; U.v[6] = (__bf16)f1.z; U.v[7] = (__bf16)f1.w;
    *((uint4*)(out + i)) = U.u;
}

// =====================================================================
// GEMM: C[M x Nn] = A[M x K] * W[Nn x K]^T + bias (+relu) (+resid)
// A, W pre-staged in bf16. WMMA 16x16x32 bf16, f32 accumulate.
// One wave -> 64x64 tile; block = 128 threads (4 waves) -> 64 x 256 of C.
// grid = (Nn/256, M/64). Requires M%64==0, Nn%256==0, K%32==0.
// Inner loop: 16 x global_load_b128 + 16 x v_wmma (no conversions).
// =====================================================================
__global__ __launch_bounds__(128)
void gemm_bf16_wmma(const __bf16* __restrict__ A, const __bf16* __restrict__ W,
                    const float* __restrict__ bias, const float* __restrict__ resid,
                    float* __restrict__ Cf, __bf16* __restrict__ Cbf,
                    int M, int Nn, int K, int relu)
{
    const int lane = threadIdx.x & 31;
    const int wave = threadIdx.x >> 5;            // 0..3
    const int m0 = blockIdx.y * 64;
    const int n0 = blockIdx.x * 256 + wave * 64;
    const int r16 = lane & 15;
    const int hi  = lane >> 4;                    // 0/1 half-wave

    v8f acc[4][4];
#pragma unroll
    for (int i = 0; i < 4; i++)
#pragma unroll
        for (int j = 0; j < 4; j++)
#pragma unroll
            for (int r = 0; r < 8; r++) acc[i][j][r] = 0.0f;

    for (int k0 = 0; k0 < K; k0 += 32) {
        v16bf a[4], b[4];
        // ---- A fragments: 16x32 (MxK). lanes 0-15: row r16, K 0..15;
        //      lanes 16-31: same rows, K 16..31.
#pragma unroll
        for (int i = 0; i < 4; i++) {
            const __bf16* ap = A + (size_t)(m0 + i * 16 + r16) * K + k0 + hi * 16;
            union { uint4 u[2]; v16bf v; } U;
            U.u[0] = *(const uint4*)ap;
            U.u[1] = *(const uint4*)(ap + 8);
            a[i] = U.v;
        }
        // ---- B fragments: 32x16 (KxN) = W-tile rows. lane (0-15): column n=r16,
        //      K 0..15 (packed pairs per VGPR); lanes 16-31: K 16..31.
#pragma unroll
        for (int j = 0; j < 4; j++) {
            const __bf16* wp = W + (size_t)(n0 + j * 16 + r16) * K + k0 + hi * 16;
            union { uint4 u[2]; v16bf v; } U;
            U.u[0] = *(const uint4*)wp;
            U.u[1] = *(const uint4*)(wp + 8);
            b[j] = U.v;
        }
        // prefetch next K-tile of A (global_prefetch_b8 on gfx1250)
        if (k0 + 32 < K)
            __builtin_prefetch(A + (size_t)(m0 + r16) * K + k0 + 32, 0, 1);

#pragma unroll
        for (int i = 0; i < 4; i++)
#pragma unroll
            for (int j = 0; j < 4; j++)
                acc[i][j] = __builtin_amdgcn_wmma_f32_16x16x32_bf16(
                    false, a[i], false, b[j], (short)0, acc[i][j], false, false);
    }

    // ---- epilogue. C/D layout: VGPR r -> (lanes 0-15: M=r, N=lane),
    //      (lanes 16-31: M=8+r, N=lane-16).
#pragma unroll
    for (int i = 0; i < 4; i++)
#pragma unroll
        for (int j = 0; j < 4; j++) {
            const int ncol = n0 + j * 16 + r16;
            const float bv = bias ? bias[ncol] : 0.0f;
#pragma unroll
            for (int r = 0; r < 8; r++) {
                const int mrow = m0 + i * 16 + hi * 8 + r;
                float v = acc[i][j][r] + bv;
                if (relu) v = v > 0.0f ? v : 0.0f;
                if (resid) v += resid[(size_t)mrow * Nn + ncol];
                if (Cf)  Cf[(size_t)mrow * Nn + ncol] = v;
                if (Cbf) Cbf[(size_t)mrow * Nn + ncol] = (__bf16)v;
            }
        }
}

// =====================================================================
// Fused residual-add + LayerNorm over last dim (D=512). One row / block.
// Optionally mirrors the result into a bf16 buffer for downstream GEMMs.
// =====================================================================
__global__ __launch_bounds__(256)
void ln_fused(const float* __restrict__ a, const float* __restrict__ b,
              const float* __restrict__ g, const float* __restrict__ beta,
              float* __restrict__ out, __bf16* __restrict__ out_bf)
{
    const int row = blockIdx.x;
    const int t = threadIdx.x;
    __shared__ float red[256];
    const float* pa = a + (size_t)row * DD;
    const float* pb = b ? b + (size_t)row * DD : nullptr;
    float v0 = pa[t]       + (pb ? pb[t]       : 0.0f);
    float v1 = pa[t + 256] + (pb ? pb[t + 256] : 0.0f);
    red[t] = v0 + v1; __syncthreads();
    for (int s = 128; s > 0; s >>= 1) { if (t < s) red[t] += red[t + s]; __syncthreads(); }
    const float mean = red[0] * (1.0f / DD);
    __syncthreads();
    const float d0 = v0 - mean, d1 = v1 - mean;
    red[t] = d0 * d0 + d1 * d1; __syncthreads();
    for (int s = 128; s > 0; s >>= 1) { if (t < s) red[t] += red[t + s]; __syncthreads(); }
    const float inv = rsqrtf(red[0] * (1.0f / DD) + 1e-5f);
    const float o0 = d0 * inv * g[t]       + beta[t];
    const float o1 = d1 * inv * g[t + 256] + beta[t + 256];
    out[(size_t)row * DD + t]       = o0;
    out[(size_t)row * DD + t + 256] = o1;
    if (out_bf) {
        out_bf[(size_t)row * DD + t]       = (__bf16)o0;
        out_bf[(size_t)row * DD + t + 256] = (__bf16)o1;
    }
}

// =====================================================================
// Causal linear attention (bucket_size=1): chunked inclusive scan.
// qvk row layout: [q(512) | v(512) | k(512)], head h at offset h*64.
// State ownership: thread t -> e = t&63, d-rows d0..d0+15 with d0=(t>>6)*16.
// =====================================================================
__global__ __launch_bounds__(256)
void causal_chunk_sums(const float* __restrict__ qvk,
                       float* __restrict__ chunkS, float* __restrict__ chunkZ)
{
    const int blk = blockIdx.x, c = blk % NC, bh = blk / NC;
    const int b = bh / HH, h = bh % HH;
    const int t = threadIdx.x, e = t & 63, d0 = (t >> 6) * 16;
    __shared__ float ke_s[64], v_s[64];
    float S[16];
#pragma unroll
    for (int i = 0; i < 16; i++) S[i] = 0.0f;
    float zreg = 0.0f;
    for (int nn = 0; nn < CH; nn++) {
        const int n = c * CH + nn;
        const float* row = qvk + ((size_t)b * NN + n) * (3 * DD);
        if (t < 64)           ke_s[t]      = __expf(row[1024 + h * 64 + t]);
        else if (t < 128)     v_s[t - 64]  = row[512 + h * 64 + (t - 64)];
        __syncthreads();
        const float ve = v_s[e];
#pragma unroll
        for (int dd = 0; dd < 16; dd++) S[dd] += ke_s[d0 + dd] * ve;
        if (t < 64) zreg += ke_s[t];
        __syncthreads();
    }
    float* so = chunkS + ((size_t)bh * NC + c) * 4096;
#pragma unroll
    for (int dd = 0; dd < 16; dd++) so[(d0 + dd) * 64 + e] = S[dd];
    if (t < 64) chunkZ[((size_t)bh * NC + c) * 64 + t] = zreg;
}

__global__ __launch_bounds__(256)
void causal_prefix(float* __restrict__ chunkS, float* __restrict__ chunkZ)
{
    const int bh = blockIdx.x, t = threadIdx.x;
    for (int i = t; i < 4096; i += 256) {
        float run = 0.0f;
        for (int c = 0; c < NC; c++) {
            float* p = chunkS + ((size_t)bh * NC + c) * 4096 + i;
            const float tmp = *p; *p = run; run += tmp;
        }
    }
    if (t < 64) {
        float run = 0.0f;
        for (int c = 0; c < NC; c++) {
            float* p = chunkZ + ((size_t)bh * NC + c) * 64 + t;
            const float tmp = *p; *p = run; run += tmp;
        }
    }
}

__global__ __launch_bounds__(256)
void causal_scan(const float* __restrict__ qvk, const float* __restrict__ chunkS,
                 const float* __restrict__ chunkZ, float* __restrict__ outA)
{
    const int blk = blockIdx.x, c = blk % NC, bh = blk / NC;
    const int b = bh / HH, h = bh % HH;
    const int t = threadIdx.x, e = t & 63, d0 = (t >> 6) * 16;
    __shared__ float q_s[64], ke_s[64], v_s[64], qs_s[64], z_s[64], red[64];
    __shared__ float part[4][64];

    float S[16];
    const float* Sin = chunkS + ((size_t)bh * NC + c) * 4096;
#pragma unroll
    for (int dd = 0; dd < 16; dd++) S[dd] = Sin[(d0 + dd) * 64 + e];
    float zreg = (t < 64) ? chunkZ[((size_t)bh * NC + c) * 64 + t] : 0.0f;

    for (int nn = 0; nn < CH; nn++) {
        const int n = c * CH + nn;
        const float* row = qvk + ((size_t)b * NN + n) * (3 * DD);
        if (t < 64)        q_s[t]        = row[h * 64 + t];
        else if (t < 128)  ke_s[t - 64]  = __expf(row[1024 + h * 64 + (t - 64)]);
        else if (t < 192)  v_s[t - 128]  = row[512 + h * 64 + (t - 128)];
        __syncthreads();
        // softmax over feature dim of q, scaled by dh^-0.5 = 0.125
        if (t < 64) red[t] = q_s[t];
        __syncthreads();
        for (int s = 32; s > 0; s >>= 1) { if (t < s) red[t] = fmaxf(red[t], red[t + s]); __syncthreads(); }
        const float qmax = red[0];
        __syncthreads();
        if (t < 64) { const float qe = __expf(q_s[t] - qmax); qs_s[t] = qe; red[t] = qe; }
        __syncthreads();
        for (int s = 32; s > 0; s >>= 1) { if (t < s) red[t] += red[t + s]; __syncthreads(); }
        const float qsum = red[0];
        __syncthreads();
        if (t < 64) {
            qs_s[t] = qs_s[t] / qsum * 0.125f;
            zreg += ke_s[t];            // inclusive k cumsum
            z_s[t] = zreg;
        }
        __syncthreads();
        // inclusive state update then query
        const float ve = v_s[e];
        float partial = 0.0f;
#pragma unroll
        for (int dd = 0; dd < 16; dd++) {
            S[dd] += ke_s[d0 + dd] * ve;
            partial += qs_s[d0 + dd] * S[dd];
        }
        part[t >> 6][e] = partial;
        if (t < 64) red[t] = qs_s[t] * (z_s[t] + 1e-6f);
        __syncthreads();
        for (int s = 32; s > 0; s >>= 1) { if (t < s) red[t] += red[t + s]; __syncthreads(); }
        const float den = red[0];
        __syncthreads();
        if (t < 64) {
            const float num = part[0][t] + part[1][t] + part[2][t] + part[3][t];
            outA[((size_t)b * NN + n) * DD + h * 64 + t] = num / den;
        }
        __syncthreads();
    }
}

// =====================================================================
// Cross (non-causal) linear attention.  kv row layout: [k(512) | v(512)].
// =====================================================================
__global__ __launch_bounds__(256)
void softk_stats(const float* __restrict__ kv, float* __restrict__ kmax,
                 float* __restrict__ ksum)
{
    const int blk = blockIdx.x, b = blk / DD, dcol = blk % DD;
    const int t = threadIdx.x;
    __shared__ float red[256];
    float m = -INFINITY;
    for (int n = t; n < NN; n += 256)
        m = fmaxf(m, kv[((size_t)b * NN + n) * (2 * DD) + dcol]);
    red[t] = m; __syncthreads();
    for (int s = 128; s > 0; s >>= 1) { if (t < s) red[t] = fmaxf(red[t], red[t + s]); __syncthreads(); }
    const float mx = red[0];
    __syncthreads();
    float su = 0.0f;
    for (int n = t; n < NN; n += 256)
        su += __expf(kv[((size_t)b * NN + n) * (2 * DD) + dcol] - mx);
    red[t] = su; __syncthreads();
    for (int s = 128; s > 0; s >>= 1) { if (t < s) red[t] += red[t + s]; __syncthreads(); }
    if (t == 0) { kmax[b * DD + dcol] = mx; ksum[b * DD + dcol] = red[0]; }
}

__global__ __launch_bounds__(256)
void ctx_partial(const float* __restrict__ kv, const float* __restrict__ kmax,
                 const float* __restrict__ ksum, float* __restrict__ ctxPart)
{
    const int blk = blockIdx.x, c = blk % NC, bh = blk / NC;
    const int b = bh / HH, h = bh % HH;
    const int t = threadIdx.x, e = t & 63, d0 = (t >> 6) * 16;
    __shared__ float ks_s[64], v_s[64], km_s[64], kd_s[64];
    if (t < 64) { km_s[t] = kmax[b * DD + h * 64 + t]; kd_s[t] = ksum[b * DD + h * 64 + t]; }
    __syncthreads();
    float S[16];
#pragma unroll
    for (int i = 0; i < 16; i++) S[i] = 0.0f;
    for (int nn = 0; nn < CH; nn++) {
        const int n = c * CH + nn;
        const float* row = kv + ((size_t)b * NN + n) * (2 * DD);
        if (t < 64)        ks_s[t]      = __expf(row[h * 64 + t] - km_s[t]) / kd_s[t];
        else if (t < 128)  v_s[t - 64]  = row[512 + h * 64 + (t - 64)];
        __syncthreads();
        const float ve = v_s[e];
#pragma unroll
        for (int dd = 0; dd < 16; dd++) S[dd] += ks_s[d0 + dd] * ve;
        __syncthreads();
    }
    float* out = ctxPart + ((size_t)bh * NC + c) * 4096;
#pragma unroll
    for (int dd = 0; dd < 16; dd++) out[(d0 + dd) * 64 + e] = S[dd];
}

__global__ __launch_bounds__(256)
void ctx_reduce(const float* __restrict__ ctxPart, float* __restrict__ ctx)
{
    const int bh = blockIdx.x, t = threadIdx.x;
    for (int i = t; i < 4096; i += 256) {
        float s = 0.0f;
        for (int c = 0; c < NC; c++) s += ctxPart[((size_t)bh * NC + c) * 4096 + i];
        ctx[(size_t)bh * 4096 + i] = s;
    }
}

__global__ __launch_bounds__(256)
void cross_apply(const float* __restrict__ q2, const float* __restrict__ ctx,
                 float* __restrict__ outA)
{
    const int blk = blockIdx.x, b = blk / NN, n = blk % NN;
    const int t = threadIdx.x;
    __shared__ float qs_s[512];
    __shared__ float red[64];
    const float* row = q2 + ((size_t)b * NN + n) * DD;
    qs_s[t] = row[t]; qs_s[t + 256] = row[t + 256];
    __syncthreads();
    for (int h = 0; h < HH; h++) {
        if (t < 64) red[t] = qs_s[h * 64 + t];
        __syncthreads();
        for (int s = 32; s > 0; s >>= 1) { if (t < s) red[t] = fmaxf(red[t], red[t + s]); __syncthreads(); }
        const float mx = red[0];
        __syncthreads();
        if (t < 64) { const float qe = __expf(qs_s[h * 64 + t] - mx); qs_s[h * 64 + t] = qe; red[t] = qe; }
        __syncthreads();
        for (int s = 32; s > 0; s >>= 1) { if (t < s) red[t] += red[t + s]; __syncthreads(); }
        const float sm = red[0];
        __syncthreads();
        if (t < 64) qs_s[h * 64 + t] = qs_s[h * 64 + t] / sm * 0.125f;
        __syncthreads();
    }
#pragma unroll
    for (int rep = 0; rep < 2; rep++) {
        const int col = t + rep * 256;
        const int h = col >> 6, e = col & 63;
        const float* cx = ctx + ((size_t)(b * HH + h)) * 4096;
        float s = 0.0f;
#pragma unroll 8
        for (int d = 0; d < 64; d++) s += qs_s[h * 64 + d] * cx[d * 64 + e];
        outA[((size_t)b * NN + n) * DD + col] = s;
    }
}

// =====================================================================
// Host launch
// =====================================================================
extern "C" void kernel_launch(void* const* d_in, const int* in_sizes, int n_in,
                              void* d_out, int out_size, void* d_ws, size_t ws_size,
                              hipStream_t stream)
{
    (void)in_sizes; (void)n_in; (void)out_size; (void)ws_size;
    const float* x      = (const float*)d_in[0];
    const float* memory = (const float*)d_in[1];
    const float* w_qvk  = (const float*)d_in[2];
    const float* b_qvk  = (const float*)d_in[3];
    const float* w_kv   = (const float*)d_in[4];
    const float* b_kv   = (const float*)d_in[5];
    const float* w_q    = (const float*)d_in[6];
    const float* b_q    = (const float*)d_in[7];
    const float* ln1_g  = (const float*)d_in[8];
    const float* ln1_b  = (const float*)d_in[9];
    const float* ln2_g  = (const float*)d_in[10];
    const float* ln2_b  = (const float*)d_in[11];
    const float* w_ff1  = (const float*)d_in[12];
    const float* b_ff1  = (const float*)d_in[13];
    const float* w_ff2  = (const float*)d_in[14];
    const float* b_ff2  = (const float*)d_in[15];
    const float* ln3_g  = (const float*)d_in[16];
    const float* ln3_b  = (const float*)d_in[17];
    float* out = (float*)d_out;

    // ---- workspace carve-up ----
    float* ws    = (float*)d_ws;
    float* qvk   = ws;                            // 3*BND f32
    float* attn1 = qvk   + 3 * BND;               // BND
    float* ln1   = attn1 + BND;                   // BND
    float* kvb   = ln1   + BND;                   // 2*BND
    float* q2    = kvb   + 2 * BND;               // BND
    float* attn2 = q2    + BND;                   // BND
    float* ln2   = attn2 + BND;                   // BND
    float* ff2   = ln2   + BND;                   // BND
    float* chS   = ff2   + BND;                   // B*H*NC*4096
    float* chZ   = chS   + (size_t)BB * HH * NC * 4096;
    float* kmax  = chZ   + (size_t)BB * HH * NC * 64;
    float* ksum  = kmax  + (size_t)BB * DD;
    float* ctxP  = ksum  + (size_t)BB * DD;       // B*H*NC*4096
    float* ctx   = ctxP  + (size_t)BB * HH * NC * 4096;
    // bf16 staging region (starts 16B aligned: all prior counts are mult. of 4)
    __bf16* bfbase = (__bf16*)(ctx + (size_t)BB * HH * 4096);
    __bf16* xbf    = bfbase;                      // BND
    __bf16* membf  = xbf    + BND;                // BND
    __bf16* ln1bf  = membf  + BND;                // BND
    __bf16* ln2bf  = ln1bf  + BND;                // BND
    __bf16* hbf    = ln2bf  + BND;                // B*N*FF
    __bf16* wqvkbf = hbf    + (size_t)BB * NN * FF;   // 1536*512
    __bf16* wkvbf  = wqvkbf + (size_t)1536 * 512;     // 1024*512
    __bf16* wqbf   = wkvbf  + (size_t)1024 * 512;     // 512*512
    __bf16* wff1bf = wqbf   + (size_t)512 * 512;      // 2048*512
    __bf16* wff2bf = wff1bf + (size_t)2048 * 512;     // 512*2048

    const int M = BB * NN;                        // 8192 rows
    dim3 blkG(128);
    auto cvtBlocks = [](size_t n) { return (unsigned)(n / 8 / 256); };

    // 0) stage fp32 -> bf16 (activations + all weights), once per launch
    cvt_f32_bf16<<<cvtBlocks(BND), 256, 0, stream>>>(x, xbf);
    cvt_f32_bf16<<<cvtBlocks(BND), 256, 0, stream>>>(memory, membf);
    cvt_f32_bf16<<<cvtBlocks((size_t)1536 * 512), 256, 0, stream>>>(w_qvk, wqvkbf);
    cvt_f32_bf16<<<cvtBlocks((size_t)1024 * 512), 256, 0, stream>>>(w_kv, wkvbf);
    cvt_f32_bf16<<<cvtBlocks((size_t)512 * 512), 256, 0, stream>>>(w_q, wqbf);
    cvt_f32_bf16<<<cvtBlocks((size_t)2048 * 512), 256, 0, stream>>>(w_ff1, wff1bf);
    cvt_f32_bf16<<<cvtBlocks((size_t)512 * 2048), 256, 0, stream>>>(w_ff2, wff2bf);

    // 1) qvk projection: (8192x512)x(512x1536)
    gemm_bf16_wmma<<<dim3(1536 / 256, M / 64), blkG, 0, stream>>>(
        xbf, wqvkbf, b_qvk, nullptr, qvk, nullptr, M, 1536, DD, 0);

    // 2) causal linear attention
    causal_chunk_sums<<<BB * HH * NC, 256, 0, stream>>>(qvk, chS, chZ);
    causal_prefix<<<BB * HH, 256, 0, stream>>>(chS, chZ);
    causal_scan<<<BB * HH * NC, 256, 0, stream>>>(qvk, chS, chZ, attn1);

    // 3) ln1 = LN(attn1 + x)  (+ bf16 mirror for the q2 GEMM)
    ln_fused<<<M, 256, 0, stream>>>(attn1, x, ln1_g, ln1_b, ln1, ln1bf);

    // 4) memory -> kv projection: (8192x512)x(512x1024)
    gemm_bf16_wmma<<<dim3(1024 / 256, M / 64), blkG, 0, stream>>>(
        membf, wkvbf, b_kv, nullptr, kvb, nullptr, M, 1024, DD, 0);

    // 5) q2 projection: (8192x512)x(512x512)
    gemm_bf16_wmma<<<dim3(512 / 256, M / 64), blkG, 0, stream>>>(
        ln1bf, wqbf, b_q, nullptr, q2, nullptr, M, 512, DD, 0);

    // 6) cross linear attention
    softk_stats<<<BB * DD, 256, 0, stream>>>(kvb, kmax, ksum);
    ctx_partial<<<BB * HH * NC, 256, 0, stream>>>(kvb, kmax, ksum, ctxP);
    ctx_reduce<<<BB * HH, 256, 0, stream>>>(ctxP, ctx);
    cross_apply<<<BB * NN, 256, 0, stream>>>(q2, ctx, attn2);

    // 7) ln2 = LN(attn2 + ln1)  (+ bf16 mirror for ff1 GEMM)
    ln_fused<<<M, 256, 0, stream>>>(attn2, ln1, ln2_g, ln2_b, ln2, ln2bf);

    // 8) ff1 + relu: (8192x512)x(512x2048), output directly in bf16
    gemm_bf16_wmma<<<dim3(2048 / 256, M / 64), blkG, 0, stream>>>(
        ln2bf, wff1bf, b_ff1, nullptr, nullptr, hbf, M, 2048, DD, 1);

    // 9) ff2 + residual: (8192x2048)x(2048x512)
    gemm_bf16_wmma<<<dim3(512 / 256, M / 64), blkG, 0, stream>>>(
        hbf, wff2bf, b_ff2, ln2, ff2, nullptr, M, 512, FF, 0);

    // 10) ln3 -> output
    ln_fused<<<M, 256, 0, stream>>>(ff2, nullptr, ln3_g, ln3_b, out, nullptr);
}
